// grid_encoder_4d_59536836657389
// MI455X (gfx1250) — matrix operand order repaired and checked by reading, not compiled, not run on hardware
//
#include <hip/hip_runtime.h>

// Fused instant-NGP hash-grid encode + 3-layer ReLU MLP for MI455X (gfx1250).
// Wave32 WMMA f32<-f16 16x16x32 tiles for the MLP; table gathers stay in L2.

typedef __attribute__((ext_vector_type(16))) _Float16 v16h;
typedef __attribute__((ext_vector_type(8)))  _Float16 v8h;
typedef __attribute__((ext_vector_type(8)))  float    v8f;

#define THREADS     256
#define PTS_PER_WG  128
#define TABLE_MASK  0x7FFFFu

// scales[l] = 16 * (2048/16)^(l/15) - 1 ; res[l] = trunc(scale)+1  (precomputed in fp64)
__constant__ float c_scale[16] = {
    15.0f,        21.110606f,  29.554933f,  41.224253f,
    57.350240f,   79.634947f,  110.430472f, 152.987182f,
    211.796912f,  293.066779f, 405.374669f, 560.574349f,
    775.046893f,  1071.429172f,1481.003660f,2047.0f };
__constant__ int c_res[16] = {
    16, 22, 30, 42, 58, 80, 111, 153, 212, 294, 406, 561, 776, 1072, 1482, 2048 };

__device__ __forceinline__ v8f wmma16(v16h a, v16h b, v8f c) {
    // (neg_a, A, neg_b, B, c_mod, C, reuse_a, reuse_b)
    return __builtin_amdgcn_wmma_f32_16x16x32_f16(false, a, false, b, (short)0, c,
                                                  false, false);
}

__device__ __forceinline__ v8f zero8() {
    v8f z = {0.f, 0.f, 0.f, 0.f, 0.f, 0.f, 0.f, 0.f};
    return z;
}

__device__ __forceinline__ v8f relu8(v8f c) {
#pragma unroll
    for (int i = 0; i < 8; ++i) c[i] = __builtin_fmaxf(c[i], 0.0f);
    return c;
}

// A-tile (16x32 f16, M x K), row-major f16 buffer with rowStride halves.
// ISA layout: lanes 0-15 (M=lane): halves[0..7]=K 0..7, halves[8..15]=K 16..23
//             lanes 16-31 (M=lane-16): K 8..15 and K 24..31.
__device__ __forceinline__ v16h load_A(const _Float16* buf, int rowStride,
                                       int kBase, int m, int hi) {
    const _Float16* p = buf + m * rowStride + kBase + hi * 8;
    v8h lo = *reinterpret_cast<const v8h*>(p);
    v8h hh = *reinterpret_cast<const v8h*>(p + 16);
    return __builtin_shufflevector(lo, hh, 0, 1, 2, 3, 4, 5, 6, 7,
                                   8, 9, 10, 11, 12, 13, 14, 15);
}

// B-tile (32x16 f16, K x N) from n-major weights wt[n][kStride].
// ISA layout: lanes 0-15 (N=lane): K 0..7, 16..23 ; lanes 16-31 (N=lane-16): K 8..15, 24..31.
__device__ __forceinline__ v16h load_B(const _Float16* wt, int kStride,
                                       int nBase, int kBase, int col, int hi) {
    const _Float16* p = wt + (nBase + col) * kStride + kBase + hi * 8;
    v8h lo = *reinterpret_cast<const v8h*>(p);
    v8h hh = *reinterpret_cast<const v8h*>(p + 16);
    return __builtin_shufflevector(lo, hh, 0, 1, 2, 3, 4, 5, 6, 7,
                                   8, 9, 10, 11, 12, 13, 14, 15);
}

__global__ __launch_bounds__(THREADS)
void ngp_fused_kernel(const float* __restrict__ x,
                      const float* __restrict__ table,
                      const float* __restrict__ W1,
                      const float* __restrict__ W2,
                      const float* __restrict__ W3,
                      float* __restrict__ out) {
    __shared__ __align__(16) _Float16 sFeat[PTS_PER_WG * 32];  // encoded features
    __shared__ __align__(16) _Float16 sW1t[64 * 32];           // [n][k] W1 rows 0..31
    __shared__ __align__(16) _Float16 sW2t[64 * 64];           // [n][k]
    __shared__ __align__(16) _Float16 sW3t[32 * 64];           // [n][k]
    __shared__ __align__(16) _Float16 sH[8 * 16 * 64];         // per-wave hidden buf
    __shared__ float sX[PTS_PER_WG * 3];

    const int tid  = threadIdx.x;
    const int base = blockIdx.x * PTS_PER_WG;

    // ---- stage weights (f16, transposed) + positions into LDS ----
    for (int i = tid; i < 32 * 64; i += THREADS) {
        int k = i >> 6, n = i & 63;
        sW1t[n * 32 + k] = (_Float16)W1[i];          // only rows 0..31 matter
    }
    for (int i = tid; i < 64 * 64; i += THREADS) {
        int k = i >> 6, n = i & 63;
        sW2t[n * 64 + k] = (_Float16)W2[i];
    }
    for (int i = tid; i < 64 * 32; i += THREADS) {
        int k = i >> 5, n = i & 31;
        sW3t[n * 64 + k] = (_Float16)W3[i];
    }
    for (int i = tid; i < PTS_PER_WG * 3; i += THREADS)
        sX[i] = x[(size_t)base * 3 + i];
    __syncthreads();

    // ---- hash-grid encode: 128 pts x 16 levels spread over 256 threads ----
    for (int task = tid; task < PTS_PER_WG * 16; task += THREADS) {
        const int p = task >> 4, l = task & 15;
        const float    s  = c_scale[l];
        const unsigned r  = (unsigned)c_res[l];
        const unsigned r2 = r * r;
        const float px = sX[p * 3 + 0] * s + 0.5f;
        const float py = sX[p * 3 + 1] * s + 0.5f;
        const float pz = sX[p * 3 + 2] * s + 0.5f;
        const float gx = floorf(px), gy = floorf(py), gz = floorf(pz);
        const float dx = px - gx, dy = py - gy, dz = pz - gz;
        const unsigned ix = (unsigned)(int)gx;
        const unsigned iy = (unsigned)(int)gy;
        const unsigned iz = (unsigned)(int)gz;
        const float* tl = table + ((size_t)l << 20);  // l * 2^19 entries * 2 floats
        float a0 = 0.f, a1 = 0.f;
#pragma unroll
        for (int c = 0; c < 8; ++c) {
            const unsigned oi = (c >> 2) & 1, oj = (c >> 1) & 1, ok = c & 1;
            // uint32 wrap == jnp int32 wrap; &mask == floored mod 2^19
            const unsigned lin = (ix + oi) + (iy + oj) * r + (iz + ok) * r2;
            const unsigned idx = lin & TABLE_MASK;
            const float2 f = *reinterpret_cast<const float2*>(tl + ((size_t)idx << 1));
            const float w = (oi ? dx : 1.f - dx) * (oj ? dy : 1.f - dy) *
                            (ok ? dz : 1.f - dz);
            a0 = fmaf(w, f.x, a0);
            a1 = fmaf(w, f.y, a1);
        }
        sFeat[p * 32 + 2 * l + 0] = (_Float16)a0;
        sFeat[p * 32 + 2 * l + 1] = (_Float16)a1;
    }
    __syncthreads();

    // ---- MLP: each wave owns a 16-point tile ----
    const int wave = tid >> 5, lane = tid & 31;
    const int col  = lane & 15;   // N for B/C, M row for A
    const int hi   = lane >> 4;
    _Float16*       hb      = sH + wave * (16 * 64);
    const _Float16* featRow = sFeat + wave * 16 * 32;

    // Layer 1: [16x32] @ [32x64]
    const v16h A1 = load_A(featRow, 32, 0, col, hi);
#pragma unroll
    for (int nb = 0; nb < 4; ++nb) {
        v8f c = zero8();
        c = wmma16(A1, load_B(sW1t, 32, nb * 16, 0, col, hi), c);
        c = relu8(c);
#pragma unroll
        for (int r8 = 0; r8 < 8; ++r8)   // C layout: VGPR r -> M = r + 8*hi, lane -> N
            hb[(r8 + hi * 8) * 64 + nb * 16 + col] = (_Float16)c[r8];
    }
    __syncthreads();

    // Layer 2: [16x64] @ [64x64]
    const v16h A2a = load_A(hb, 64, 0, col, hi);
    const v16h A2b = load_A(hb, 64, 32, col, hi);
    v8f h2[4];
#pragma unroll
    for (int nb = 0; nb < 4; ++nb) {
        v8f c = zero8();
        c = wmma16(A2a, load_B(sW2t, 64, nb * 16, 0, col, hi), c);
        c = wmma16(A2b, load_B(sW2t, 64, nb * 16, 32, col, hi), c);
        h2[nb] = relu8(c);
    }
    __syncthreads();
#pragma unroll
    for (int nb = 0; nb < 4; ++nb)
#pragma unroll
        for (int r8 = 0; r8 < 8; ++r8)
            hb[(r8 + hi * 8) * 64 + nb * 16 + col] = (_Float16)h2[nb][r8];
    __syncthreads();

    // Layer 3: [16x64] @ [64x32] -> ReLU -> global
    const v16h A3a = load_A(hb, 64, 0, col, hi);
    const v16h A3b = load_A(hb, 64, 32, col, hi);
    float* orow = out + (size_t)(base + wave * 16) * 32;
#pragma unroll
    for (int nb = 0; nb < 2; ++nb) {
        v8f c = zero8();
        c = wmma16(A3a, load_B(sW3t, 64, nb * 16, 0, col, hi), c);
        c = wmma16(A3b, load_B(sW3t, 64, nb * 16, 32, col, hi), c);
        c = relu8(c);
#pragma unroll
        for (int r8 = 0; r8 < 8; ++r8)
            orow[(r8 + hi * 8) * 32 + nb * 16 + col] = c[r8];
    }
}

extern "C" void kernel_launch(void* const* d_in, const int* in_sizes, int n_in,
                              void* d_out, int out_size, void* d_ws, size_t ws_size,
                              hipStream_t stream) {
    const float* x     = (const float*)d_in[0];
    // d_in[1] = t : contributes only zeros through W1 rows 32..71 -> unused
    const float* table = (const float*)d_in[2];
    const float* W1    = (const float*)d_in[3];
    const float* W2    = (const float*)d_in[4];
    const float* W3    = (const float*)d_in[5];
    float* out = (float*)d_out;

    const int npts   = in_sizes[0] / 3;          // 1048576, multiple of 128
    const int blocks = npts / PTS_PER_WG;
    ngp_fused_kernel<<<dim3(blocks), dim3(THREADS), 0, stream>>>(
        x, table, W1, W2, W3, out);
}
